// Memory_23184233464642
// MI455X (gfx1250) — compile-verified, matrix operand order checked by reading, not run on hardware
//
#include <hip/hip_runtime.h>
#include <hip/hip_bf16.h>

// ---------------------------------------------------------------------------
// CDNA5 (gfx1250) implementation of the memory-attention reference.
// GEMMs use v_wmma_f32_16x16x32_bf16 with a bf16 hi/lo split ("bf16x3") to
// retain ~fp32 precision (needed so top-k selection matches fp32 ordering)
// while running on the fast bf16 WMMA path. wave32, 8 waves / 256 threads.
// ---------------------------------------------------------------------------

typedef __attribute__((ext_vector_type(16))) __bf16 v16bf;
typedef __attribute__((ext_vector_type(8)))  __bf16 v8bf;
typedef __attribute__((ext_vector_type(8)))  float  v8f;

#define TLDS 40   // LDS row stride in bf16 elems (32 + 8 pad, keeps 16B align)

__device__ __forceinline__ void split_bf16(float f, __bf16& h, __bf16& l) {
  union { unsigned u; float f; } a; a.f = f;
  unsigned hu = a.u & 0xFFFF0000u;
  union { unsigned u; float f; } hf; hf.u = hu;
  float r = f - hf.f;
  union { unsigned u; float f; } rb; rb.f = r;
  union { unsigned short s; __bf16 b; } c0, c1;
  c0.s = (unsigned short)(hu >> 16);
  c1.s = (unsigned short)(rb.u >> 16);
  h = c0.b; l = c1.b;
}

// Load one 16x32 bf16 A/B fragment half per lane: K = base..base+7 and
// 16+base..16+base+7, contiguous in LDS row-major [row][k] tiles.
__device__ __forceinline__ v16bf load_frag(const __bf16* p) {
  v8bf a = *(const v8bf*)p;
  v8bf b = *(const v8bf*)(p + 16);
  return __builtin_shufflevector(a, b, 0,1,2,3,4,5,6,7,8,9,10,11,12,13,14,15);
}

// C = A(MxK,f32) * B + [bias] + [residual], [relu]
// BMODE 0: B is [N][K] row-major (C = A*B^T, e.g. mem_c)
// BMODE 1: B is [K][N] row-major (C = A*W,   e.g. weights)
// Block tile 128x128, K-step 32, 8 waves; wave tile 64(M) x 32(N).
template<int BMODE, bool BIAS, bool RELU, bool RES>
__global__ __launch_bounds__(256) void gemm_bf16x3(
    const float* __restrict__ A, int lda, int Kact,
    const float* __restrict__ B, int ldb,
    const float* __restrict__ bias,
    const float* __restrict__ resid,
    float* __restrict__ C, long ldc, int Nact)
{
  __shared__ __bf16 As_h[128 * TLDS];
  __shared__ __bf16 As_l[128 * TLDS];
  __shared__ __bf16 Bs_h[128 * TLDS];
  __shared__ __bf16 Bs_l[128 * TLDS];

  const int tid  = threadIdx.x;
  const int lane = tid & 31;
  const int wave = tid >> 5;
  const int wm   = wave & 1;   // 2 M groups of 64 rows
  const int wn   = wave >> 1;  // 4 N groups of 32 cols
  const int m0   = blockIdx.y * 128;
  const int n0   = blockIdx.x * 128;
  const int KT   = (Kact + 31) & ~31;

  v8f acc[4][2];
  #pragma unroll
  for (int i = 0; i < 4; ++i)
    #pragma unroll
    for (int j = 0; j < 2; ++j)
      #pragma unroll
      for (int r = 0; r < 8; ++r) acc[i][j][r] = 0.f;

  for (int k0 = 0; k0 < KT; k0 += 32) {
    // ---- stage A tile (128x32) as hi/lo bf16 in LDS ----
    #pragma unroll
    for (int i = 0; i < 16; ++i) {
      int e = tid + i * 256;
      int r = e >> 5, cc = e & 31;
      int kk = k0 + cc;
      float f = (kk < Kact) ? A[(long)(m0 + r) * lda + kk] : 0.f;
      __bf16 h, l; split_bf16(f, h, l);
      As_h[r * TLDS + cc] = h;
      As_l[r * TLDS + cc] = l;
    }
    // ---- stage B tile (128 n x 32 k), stored [n][k] in LDS ----
    if (BMODE == 0) {
      #pragma unroll
      for (int i = 0; i < 16; ++i) {
        int e = tid + i * 256;
        int r = e >> 5, cc = e & 31;
        int gn = n0 + r, gk = k0 + cc;
        float f = (gn < Nact && gk < Kact) ? B[(long)gn * ldb + gk] : 0.f;
        __bf16 h, l; split_bf16(f, h, l);
        Bs_h[r * TLDS + cc] = h;
        Bs_l[r * TLDS + cc] = l;
      }
    } else {
      #pragma unroll
      for (int i = 0; i < 16; ++i) {
        int e = tid + i * 256;
        int nn = e & 127, kk = e >> 7;
        int gn = n0 + nn, gk = k0 + kk;
        float f = (gn < Nact && gk < Kact) ? B[(long)gk * ldb + gn] : 0.f;
        __bf16 h, l; split_bf16(f, h, l);
        Bs_h[nn * TLDS + kk] = h;
        Bs_l[nn * TLDS + kk] = l;
      }
    }
    __syncthreads();

    // ---- WMMA: bf16x3 ----
    const int bk = (lane >> 4) * 8;   // K base per half-wave
    const int mr = lane & 15;
    v16bf ah[4], al[4];
    #pragma unroll
    for (int mt = 0; mt < 4; ++mt) {
      int row = wm * 64 + mt * 16 + mr;
      ah[mt] = load_frag(&As_h[row * TLDS + bk]);
      al[mt] = load_frag(&As_l[row * TLDS + bk]);
    }
    #pragma unroll
    for (int nt = 0; nt < 2; ++nt) {
      int col = wn * 32 + nt * 16 + mr;
      v16bf bh = load_frag(&Bs_h[col * TLDS + bk]);
      v16bf bl = load_frag(&Bs_l[col * TLDS + bk]);
      #pragma unroll
      for (int mt = 0; mt < 4; ++mt) {
        acc[mt][nt] = __builtin_amdgcn_wmma_f32_16x16x32_bf16(
            false, ah[mt], false, bh, (short)0, acc[mt][nt], false, false);
        acc[mt][nt] = __builtin_amdgcn_wmma_f32_16x16x32_bf16(
            false, ah[mt], false, bl, (short)0, acc[mt][nt], false, false);
        acc[mt][nt] = __builtin_amdgcn_wmma_f32_16x16x32_bf16(
            false, al[mt], false, bh, (short)0, acc[mt][nt], false, false);
      }
    }
    __syncthreads();
  }

  // ---- epilogue: C/D layout VGPR r -> (M = r + 8*(lane>>4), N = lane&15) ----
  const int rsel = (lane >> 4) * 8;
  #pragma unroll
  for (int mt = 0; mt < 4; ++mt) {
    #pragma unroll
    for (int nt = 0; nt < 2; ++nt) {
      int col = n0 + wn * 32 + nt * 16 + (lane & 15);
      if (col < Nact) {
        int rbase = m0 + wm * 64 + mt * 16 + rsel;
        #pragma unroll
        for (int r = 0; r < 8; ++r) {
          float v = acc[mt][nt][r];
          if (BIAS) v += bias[col];
          long off = (long)(rbase + r) * ldc + col;
          if (RES) v += resid[off];
          if (RELU) v = fmaxf(v, 0.f);
          C[off] = v;
        }
      }
    }
  }
}

// ---------------------------------------------------------------------------
// Exact top-64 per row (histogram select + bitonic sort of candidates).
// ---------------------------------------------------------------------------
__device__ __forceinline__ unsigned mono_key(float f) {
  unsigned u = __float_as_uint(f);
  return (u & 0x80000000u) ? ~u : (u | 0x80000000u);
}
__device__ __forceinline__ float inv_key(unsigned k) {
  unsigned u = (k & 0x80000000u) ? (k & 0x7FFFFFFFu) : ~k;
  return __uint_as_float(u);
}

__global__ __launch_bounds__(256) void topk_kernel(
    const float* __restrict__ tau, float* __restrict__ dvals,
    int* __restrict__ didx, int N)
{
  __shared__ unsigned hist[4096];
  __shared__ unsigned long long cand[2048];
  __shared__ unsigned s_cnt;
  __shared__ int s_bin;
  const int tid = threadIdx.x;
  const long row = blockIdx.x;
  const float* tr = tau + row * N;

  for (int i = tid; i < 4096; i += 256) hist[i] = 0u;
  __syncthreads();
  for (int i = tid; i < N; i += 256)
    atomicAdd(&hist[mono_key(tr[i]) >> 20], 1u);
  __syncthreads();
  if (tid == 0) {
    unsigned cum = 0; int b = 4095;
    for (; b > 0; --b) { cum += hist[b]; if (cum >= 64u) break; }
    s_bin = b; s_cnt = 0u;
  }
  __syncthreads();
  const unsigned bstar = (unsigned)s_bin;
  for (int i = tid; i < N; i += 256) {
    unsigned key = mono_key(tr[i]);
    if ((key >> 20) >= bstar) {
      unsigned pos = atomicAdd(&s_cnt, 1u);
      if (pos < 2048u)
        cand[pos] = ((unsigned long long)key << 32) | (unsigned)(~(unsigned)i);
    }
  }
  __syncthreads();
  unsigned total = s_cnt; if (total > 2048u) total = 2048u;
  unsigned P = 64u; while (P < total) P <<= 1;
  for (unsigned e = total + tid; e < P; e += 256) cand[e] = 0ull;
  __syncthreads();
  for (unsigned ssz = 2; ssz <= P; ssz <<= 1) {
    for (unsigned str = ssz >> 1; str > 0; str >>= 1) {
      for (unsigned e = tid; e < P; e += 256) {
        unsigned p = e ^ str;
        if (p > e) {
          unsigned long long x = cand[e], y = cand[p];
          bool up = ((e & ssz) == 0);
          if (up ? (x < y) : (x > y)) { cand[e] = y; cand[p] = x; }
        }
      }
      __syncthreads();
    }
  }
  if (tid < 64) {
    unsigned long long kv = cand[tid];
    dvals[row * 64 + tid] = inv_key((unsigned)(kv >> 32));
    didx [row * 64 + tid] = (int)(~(unsigned)kv);
  }
}

// mems[row] = [delta, mem_c[idx][0..510]]
__global__ __launch_bounds__(256) void build_mems_kernel(
    const float* __restrict__ dvals, const int* __restrict__ didx,
    const float* __restrict__ mem_c, float* __restrict__ mems)
{
  const long row = blockIdx.x;
  const int tid = threadIdx.x;
  const long src = didx[row];
  float* o = mems + row * 512;
  if (tid == 0) o[0] = dvals[row];
  const float* s = mem_c + src * 511;
  for (int c = tid; c < 511; c += 256) o[1 + c] = s[c];
}

// out = LN(x [+ res]) * g + b ; one wave32 per row, 8 rows per block
__global__ __launch_bounds__(256) void ln_kernel(
    const float* __restrict__ x, const float* __restrict__ res,
    const float* __restrict__ g, const float* __restrict__ bta,
    float* __restrict__ out, int D)
{
  const int lane = threadIdx.x & 31;
  const long row = (long)blockIdx.x * 8 + (threadIdx.x >> 5);
  const float* xr = x + row * D;
  const float* rr = res ? res + row * D : nullptr;
  float s = 0.f, s2 = 0.f;
  for (int c = lane; c < D; c += 32) {
    float v = xr[c]; if (res) v += rr[c];
    s += v; s2 += v * v;
  }
  for (int m = 16; m; m >>= 1) { s += __shfl_xor(s, m, 32); s2 += __shfl_xor(s2, m, 32); }
  float mu = s / D;
  float var = s2 / D - mu * mu;
  float rs = rsqrtf(var + 1e-5f);
  float* o = out + row * D;
  for (int c = lane; c < D; c += 32) {
    float v = xr[c]; if (res) v += rr[c];
    o[c] = (v - mu) * rs * g[c] + bta[c];
  }
}

// per-batch 64x64 attention over qkv rows (q|k|v = 64|64|512 of 640)
__global__ __launch_bounds__(256) void attn_kernel(
    const float* __restrict__ qkv, float* __restrict__ att)
{
  __shared__ float sq[64 * 65];
  __shared__ float sk[64 * 65];
  __shared__ float sw[64 * 65];
  const int tid = threadIdx.x;
  const long b = blockIdx.x;
  const float* base = qkv + b * 64 * 640;
  for (int e = tid; e < 4096; e += 256) {
    int r = e >> 6, c = e & 63;
    sq[r * 65 + c] = base[(long)r * 640 + c];
    sk[r * 65 + c] = base[(long)r * 640 + 64 + c];
  }
  __syncthreads();
  for (int e = tid; e < 4096; e += 256) {
    int i = e >> 6, j = e & 63;
    float s = 0.f;
    for (int c = 0; c < 64; ++c) s += sq[i * 65 + c] * sk[j * 65 + c];
    sw[i * 65 + j] = s * 0.125f;   // key_size^-0.5 folded into scores
  }
  __syncthreads();
  if (tid < 64) {
    float m = -3.4e38f;
    for (int j = 0; j < 64; ++j) m = fmaxf(m, sw[tid * 65 + j]);
    float sum = 0.f;
    for (int j = 0; j < 64; ++j) { float e2 = expf(sw[tid * 65 + j] - m); sw[tid * 65 + j] = e2; sum += e2; }
    float inv = 1.f / sum;
    for (int j = 0; j < 64; ++j) sw[tid * 65 + j] *= inv;
  }
  __syncthreads();
  for (int e = tid; e < 64 * 512; e += 256) {
    int i = e >> 9, d = e & 511;
    float s = 0.f;
    for (int j = 0; j < 64; ++j) s += sw[i * 65 + j] * base[(long)j * 640 + 128 + d];
    att[(b * 64 + i) * 512 + d] = s;
  }
}

// ---------------------------------------------------------------------------
extern "C" void kernel_launch(void* const* d_in, const int* in_sizes, int n_in,
                              void* d_out, int out_size, void* d_ws, size_t ws_size,
                              hipStream_t stream) {
  const float* c_in  = (const float*)d_in[0];
  const float* mem_c = (const float*)d_in[1];
  const float* Wqkv  = (const float*)d_in[2];
  const float* bqkv  = (const float*)d_in[3];
  const float* g_qkv = (const float*)d_in[4];
  const float* b_qkv = (const float*)d_in[5];
  const float* g_mem = (const float*)d_in[6];
  const float* b_mem = (const float*)d_in[7];
  const float* W1  = (const float*)d_in[8];
  const float* b1  = (const float*)d_in[9];
  const float* W2  = (const float*)d_in[10];
  const float* b2  = (const float*)d_in[11];
  const float* Ws1 = (const float*)d_in[12];
  const float* bs1 = (const float*)d_in[13];
  const float* Ws2 = (const float*)d_in[14];
  const float* bs2 = (const float*)d_in[15];
  (void)in_sizes; (void)n_in; (void)out_size; (void)ws_size;

  // workspace layout (tau region reused for qkv + attended after top-k)
  char* w = (char*)d_ws;
  const size_t TAU_SZ  = 256ull * 100000ull * 4ull;  // 102,400,000
  const size_t MEM_SZ  = 16384ull * 512ull * 4ull;   //  33,554,432
  const size_t QKV_SZ  = 16384ull * 640ull * 4ull;   //  41,943,040
  float* tau    = (float*)w;
  float* qkv    = (float*)w;                 // alias (after top-k)
  float* att    = (float*)(w + QKV_SZ);      // alias (after top-k)
  char*  r2     = w + TAU_SZ;
  float* deltas = (float*)r2;
  int*   idxbuf = (int*)(r2 + 65536);
  float* mems   = (float*)(r2 + 131072);
  float* memA   = (float*)(r2 + 131072 + MEM_SZ);
  float* memB   = (float*)(r2 + 131072 + 2 * MEM_SZ);
  float* hbuf   = (float*)(r2 + 131072 + 3 * MEM_SZ);
  float* outp   = (float*)d_out;

  dim3 blk(256);

  // 1) tau[256,100000] = c @ mem_c^T
  gemm_bf16x3<0, false, false, false><<<dim3(782, 2), blk, 0, stream>>>(
      c_in, 511, 511, mem_c, 511, nullptr, nullptr, tau, 100000L, 100000);

  // 2) exact top-64 per row
  topk_kernel<<<256, blk, 0, stream>>>(tau, deltas, idxbuf, 100000);

  // 3) mems[16384,512] = [delta | gathered row]
  build_mems_kernel<<<16384, blk, 0, stream>>>(deltas, idxbuf, mem_c, mems);

  // 4) qkv = LN(mems @ Wqkv + bqkv)
  gemm_bf16x3<1, true, false, false><<<dim3(5, 128), blk, 0, stream>>>(
      mems, 512, 512, Wqkv, 640, bqkv, nullptr, qkv, 640L, 640);
  ln_kernel<<<2048, blk, 0, stream>>>(qkv, nullptr, g_qkv, b_qkv, qkv, 640);

  // 5) attention
  attn_kernel<<<256, blk, 0, stream>>>(qkv, att);

  // 6) mem = LN(mems + attended)
  ln_kernel<<<2048, blk, 0, stream>>>(mems, att, g_mem, b_mem, memA, 512);

  // 7) two residual MLP blocks: mem = LN(relu(mem@W1+b1)@W2 + b2 + mem)
  for (int it = 0; it < 2; ++it) {
    gemm_bf16x3<1, true, true, false><<<dim3(4, 128), blk, 0, stream>>>(
        memA, 512, 512, W1, 512, b1, nullptr, hbuf, 512L, 512);
    gemm_bf16x3<1, true, false, true><<<dim3(4, 128), blk, 0, stream>>>(
        hbuf, 512, 512, W2, 512, b2, memA, memB, 512L, 512);
    ln_kernel<<<2048, blk, 0, stream>>>(memB, nullptr, g_mem, b_mem, memA, 512);
  }

  // 8) c_prime = relu(mem@Ws1+bs1)@Ws2 + bs2  -> d_out (16384 x 511)
  gemm_bf16x3<1, true, true, false><<<dim3(4, 128), blk, 0, stream>>>(
      memA, 512, 512, Ws1, 512, bs1, nullptr, hbuf, 512L, 512);
  gemm_bf16x3<1, true, false, false><<<dim3(4, 128), blk, 0, stream>>>(
      hbuf, 512, 512, Ws2, 511, bs2, nullptr, outp, 511L, 511);
}
